// MultiheadAttention_18897856102840
// MI455X (gfx1250) — matrix operand order repaired and checked by reading, not compiled, not run on hardware
//
#include <hip/hip_runtime.h>
#include <hip/hip_bf16.h>

// ---------------------------------------------------------------------------
// CDNA5 (gfx1250) MHA forward: bf16 WMMA GEMMs + fused flash attention.
// Wave32, V_WMMA_F32_16X16X32_BF16 (16x16x32, f32 accum).
// K-tiles in attention staged via the Tensor Data Mover (TENSOR_LOAD_TO_LDS,
// tracked with TENSORcnt) to keep the SIMDs free for WMMA issue.
// ---------------------------------------------------------------------------

typedef __attribute__((ext_vector_type(16))) __bf16 v16bf;
typedef __attribute__((ext_vector_type(8)))  float  v8f;
typedef __attribute__((ext_vector_type(4)))  unsigned int u32x4;
typedef __attribute__((ext_vector_type(8)))  int   i32x8;
typedef __attribute__((ext_vector_type(4)))  int   i32x4;

union FragU { v16bf v; uint4 q[2]; };

__device__ __forceinline__ unsigned short f2bf(float f) {
  union { float f; unsigned u; } x; x.f = f;
  unsigned r = x.u + 0x7FFFu + ((x.u >> 16) & 1u);   // round-to-nearest-even
  return (unsigned short)(r >> 16);
}

__device__ __forceinline__ v8f wmma_bf16(v16bf a, v16bf b, v8f c) {
  // (neg_a, A, neg_b, B, c_mod, C, reuse_a, reuse_b)
  return __builtin_amdgcn_wmma_f32_16x16x32_bf16(false, a, false, b, (short)0, c,
                                                 false, false);
}

// A fragment (16x32 bf16, M x K). Per ISA 7.12.2:
// lane m=lane&15, half=lane>>4: V0-3 hold K = half*8+0..7, V4-7 hold K = 16+half*8+0..7.
__device__ __forceinline__ v16bf frag_a(const unsigned short* s, int ld, int lane) {
  const int m = lane & 15, hl = lane >> 4;
  FragU f;
  f.q[0] = *(const uint4*)(s + m * ld + hl * 8);
  f.q[1] = *(const uint4*)(s + m * ld + 16 + hl * 8);
  return f.v;
}

// B fragment (32x16 bf16, K x N), read from a K-major (transposed) tile:
// lane n=lane&15, half=lane>>4: V0-3 hold K = half*8+0..7, V4-7 hold K = 16+half*8+0..7.
__device__ __forceinline__ v16bf frag_b(const unsigned short* s, int ld, int lane) {
  const int n = lane & 15, hl = lane >> 4;
  FragU f;
  f.q[0] = *(const uint4*)(s + n * ld + hl * 8);
  f.q[1] = *(const uint4*)(s + n * ld + 16 + hl * 8);
  return f.v;
}

// ---------------------------------------------------------------------------
// Tensor Data Mover: DMA a 64x64 bf16 tile (row stride 64 elems) from global
// memory into LDS. D# layout per CDNA5 ISA.
// Issue from ONE wave only (TDM ignores EXEC); completion via TENSORcnt.
// ---------------------------------------------------------------------------
__device__ __forceinline__ void tdm_load_tile_64x64_bf16(const unsigned short* gsrc,
                                                         unsigned ldsAddr,
                                                         unsigned tensorRows) {
  const unsigned long long ga = (unsigned long long)(uintptr_t)gsrc;
  u32x4 g0;
  g0[0] = 1u;                                   // count=1, no gather, user D#
  g0[1] = ldsAddr;                              // lds_addr[31:0]
  g0[2] = (unsigned)(ga & 0xFFFFFFFFu);         // global_addr[31:0]
  g0[3] = (unsigned)((ga >> 32) & 0x01FFFFFFu)  // global_addr[56:32]
          | (2u << 30);                         // type = 2 ("image")
  i32x8 g1;
  g1[0] = (int)(1u << 16);                      // data_size=1 -> 2 bytes
  g1[1] = (int)(64u << 16);                     // tensor_dim0[15:0]=64 @bit48
  g1[2] = (int)(tensorRows << 16);              // dim0 hi=0 | tensor_dim1 lo @bit80
  g1[3] = (int)((tensorRows >> 16) | (64u << 16)); // tensor_dim1 hi | tile_dim0=64
  g1[4] = (int)64u;                             // tile_dim1=64 | tile_dim2=0
  g1[5] = (int)64u;                             // tensor_dim0_stride[31:0]=64
  g1[6] = 0;                                    // stride hi | tensor_dim1_stride lo
  g1[7] = 0;
  i32x4 gz = (i32x4){0, 0, 0, 0};               // groups 2/3 unused (2D tile)
#if defined(__clang_major__) && __clang_major__ >= 23
  __builtin_amdgcn_tensor_load_to_lds(g0, g1, gz, gz, (i32x8){0,0,0,0,0,0,0,0}, 0);
#else
  __builtin_amdgcn_tensor_load_to_lds(g0, g1, gz, gz, 0);
#endif
}

// ---------------------------------------------------------------------------
// GEMM: Y[M,N] = A[M,K] @ W[K,N] (+ bias). A fp32 or bf16; W fp32, converted.
// 128x128 block tile, BK=32, 8 waves (4x2), each wave 32x64 = 2x4 WMMA accs.
// ---------------------------------------------------------------------------
template <bool A_BF16, bool ADD_BIAS>
__global__ __launch_bounds__(256) void gemm_ln_proj(const void* __restrict__ Aip,
                                                    const float* __restrict__ W,
                                                    const float* __restrict__ bias,
                                                    float* __restrict__ Y,
                                                    int M, int N, int K) {
  __shared__ __align__(16) unsigned short As[128 * 32];
  __shared__ __align__(16) unsigned short Bt[128 * 32];  // W tile, transposed (n-major)

  const int tid  = threadIdx.x;
  const int lane = tid & 31, wid = tid >> 5;
  const int wm = wid >> 1, wn = wid & 1;
  const int m0 = blockIdx.y * 128, n0 = blockIdx.x * 128;
  const int nl = lane & 15, hl = lane >> 4;

  v8f acc[2][4];
#pragma unroll
  for (int i = 0; i < 2; ++i)
#pragma unroll
    for (int t = 0; t < 4; ++t) acc[i][t] = (v8f){0.f,0.f,0.f,0.f,0.f,0.f,0.f,0.f};

  const int ar = tid >> 1, aks = (tid & 1) * 16;   // A stage: 16 elems/thread
  const int bk = tid & 31, bns = (tid >> 5) * 16;  // W stage: 16 elems/thread

  for (int k0 = 0; k0 < K; k0 += 32) {
    // ---- stage A tile (128x32) as bf16, row-major ----
    if (A_BF16) {
      const unsigned short* p =
          (const unsigned short*)Aip + (size_t)(m0 + ar) * K + k0 + aks;
      __builtin_prefetch(p + 32, 0, 1);
      *(uint4*)(As + ar * 32 + aks)     = *(const uint4*)p;
      *(uint4*)(As + ar * 32 + aks + 8) = *(const uint4*)(p + 8);
    } else {
      const float* p = (const float*)Aip + (size_t)(m0 + ar) * K + k0 + aks;
      __builtin_prefetch(p + 32, 0, 1);
#pragma unroll
      for (int j = 0; j < 16; ++j) As[ar * 32 + aks + j] = f2bf(p[j]);
    }
    // ---- stage W tile (32x128) transposed -> Bt[n][k] ----
    {
      const float* p = W + (size_t)(k0 + bk) * N + n0 + bns;
#pragma unroll
      for (int j = 0; j < 16; ++j) Bt[(bns + j) * 32 + bk] = f2bf(p[j]);
    }
    __syncthreads();

    v16bf aF[2], bF[4];
#pragma unroll
    for (int i = 0; i < 2; ++i) aF[i] = frag_a(As + (wm * 32 + i * 16) * 32, 32, lane);
#pragma unroll
    for (int t = 0; t < 4; ++t) bF[t] = frag_b(Bt + (wn * 64 + t * 16) * 32, 32, lane);
#pragma unroll
    for (int i = 0; i < 2; ++i)
#pragma unroll
      for (int t = 0; t < 4; ++t) acc[i][t] = wmma_bf16(aF[i], bF[t], acc[i][t]);
    __syncthreads();
  }

  // Epilogue. C layout: VGPR r -> row r (lanes 0-15) / row r+8 (lanes 16-31).
#pragma unroll
  for (int i = 0; i < 2; ++i)
#pragma unroll
    for (int t = 0; t < 4; ++t) {
      const int col = n0 + wn * 64 + t * 16 + nl;
      const float bval = ADD_BIAS ? bias[col] : 0.0f;
#pragma unroll
      for (int r = 0; r < 8; ++r) {
        const int row = m0 + wm * 32 + i * 16 + hl * 8 + r;
        Y[(size_t)row * N + col] = acc[i][t][r] + bval;
      }
    }
}

// ---------------------------------------------------------------------------
// LayerNorm (optional) + permute to head-major bf16 [B,H,S,64].
// One block per (b,s) row of 1024; outscale folds attention's 1/sqrt(D) into Q.
// ---------------------------------------------------------------------------
template <bool DO_LN>
__global__ __launch_bounds__(256) void ln_permute(const float* __restrict__ X,
                                                  const float* __restrict__ gamma,
                                                  const float* __restrict__ beta,
                                                  unsigned short* __restrict__ out,
                                                  float outscale) {
  __shared__ float s1[8], s2[8];
  const int row = blockIdx.x;                 // b*2048 + s
  const int b = row >> 11, s = row & 2047;
  const int tid = threadIdx.x;

  const float4 xv = *(const float4*)(X + (size_t)row * 1024 + tid * 4);
  const float* xa = &xv.x;

  float mu = 0.f, rs = 1.f;
  if (DO_LN) {
    float sum = xv.x + xv.y + xv.z + xv.w;
    float sq  = xv.x * xv.x + xv.y * xv.y + xv.z * xv.z + xv.w * xv.w;
#pragma unroll
    for (int off = 16; off > 0; off >>= 1) {
      sum += __shfl_xor(sum, off, 32);
      sq  += __shfl_xor(sq,  off, 32);
    }
    if ((tid & 31) == 0) { s1[tid >> 5] = sum; s2[tid >> 5] = sq; }
    __syncthreads();
    float tot = 0.f, totq = 0.f;
#pragma unroll
    for (int w = 0; w < 8; ++w) { tot += s1[w]; totq += s2[w]; }
    mu = tot * (1.0f / 1024.0f);
    const float var = totq * (1.0f / 1024.0f) - mu * mu;
    rs = rsqrtf(var + 1e-5f);
  }

#pragma unroll
  for (int j = 0; j < 4; ++j) {
    const int idx = tid * 4 + j;
    float v = xa[j];
    if (DO_LN) v = (v - mu) * rs * gamma[idx] + beta[idx];
    v *= outscale;
    const int h = idx >> 6, d = idx & 63;
    out[(((size_t)b * 16 + h) * 2048 + s) * 64 + d] = f2bf(v);
  }
}

// ---------------------------------------------------------------------------
// Flash attention, one block per (b, h, 64-query tile). 4 waves x 16 rows.
// Bc = 64 keys per iteration; hd = 64. Online softmax, O/l/m in registers.
// K tile staged by the Tensor Data Mover; V staged transposed by VALU path.
// Output written bf16 into [B,S,D] layout (col = h*64 + d) for the Wo GEMM.
// ---------------------------------------------------------------------------
__global__ __launch_bounds__(128) void flash_attn(const unsigned short* __restrict__ Q,
                                                  const unsigned short* __restrict__ Kd,
                                                  const unsigned short* __restrict__ V,
                                                  unsigned short* __restrict__ O) {
  __shared__ __align__(16) unsigned short Qs[64 * 64];
  __shared__ __align__(16) unsigned short Ks[64 * 64];
  __shared__ __align__(16) unsigned short Vt[64 * 64];  // V transposed: Vt[d][k]
  __shared__ __align__(16) unsigned short Ps[64 * 64];  // softmax numerators

  const int tid = threadIdx.x, lane = tid & 31, wid = tid >> 5;
  const int nl = lane & 15, hl = lane >> 4;
  const int qt = blockIdx.x, h = blockIdx.y, b = blockIdx.z;
  const int s0 = qt * 64;
  const size_t headOff = (((size_t)b * 16 + h) * 2048) * 64;
  const unsigned short* Qh = Q  + headOff;
  const unsigned short* Kh = Kd + headOff;
  const unsigned short* Vh = V  + headOff;
  const unsigned KsLds = (unsigned)(uintptr_t)(&Ks[0]);  // low 32 bits = LDS addr

  const int r  = tid >> 1;          // staging row 0..63
  const int cs = (tid & 1) * 32;    // staging column segment

  // stage Q tile once (scale already folded into Q by ln_permute)
  {
    const unsigned short* p = Qh + (size_t)(s0 + r) * 64 + cs;
    *(uint4*)(Qs + r * 64 + cs)      = *(const uint4*)p;
    *(uint4*)(Qs + r * 64 + cs + 8)  = *(const uint4*)(p + 8);
    *(uint4*)(Qs + r * 64 + cs + 16) = *(const uint4*)(p + 16);
    *(uint4*)(Qs + r * 64 + cs + 24) = *(const uint4*)(p + 24);
  }

  v8f oacc[4];
#pragma unroll
  for (int t = 0; t < 4; ++t) oacc[t] = (v8f){0.f,0.f,0.f,0.f,0.f,0.f,0.f,0.f};
  float mrun[8], lrun[8];
#pragma unroll
  for (int i = 0; i < 8; ++i) { mrun[i] = -3.0e38f; lrun[i] = 0.0f; }

  const int qr = wid * 16;  // this wave's query-row base within the tile

  for (int kb = 0; kb < 2048; kb += 64) {
    // ---- stage K tile via Tensor Data Mover (wave 0 only; EXEC-independent) ----
    if (wid == 0) {
      tdm_load_tile_64x64_bf16(Kh + (size_t)kb * 64, KsLds, 2048u);
      __builtin_amdgcn_s_wait_tensorcnt(0);
    }
    // ---- stage V tile transposed (all waves) ----
    {
      const unsigned short* p = Vh + (size_t)(kb + r) * 64 + cs;
#pragma unroll
      for (int j = 0; j < 32; ++j) Vt[(cs + j) * 64 + r] = p[j];
    }
    __syncthreads();

    // S = Q K^T : A = Q rows, B = K^T (== K rows, row-contiguous in d)
    const v16bf qa0 = frag_a(Qs + qr * 64,      64, lane);
    const v16bf qa1 = frag_a(Qs + qr * 64 + 32, 64, lane);
    v8f sacc[4];
#pragma unroll
    for (int t = 0; t < 4; ++t) {
      v8f sz = (v8f){0.f,0.f,0.f,0.f,0.f,0.f,0.f,0.f};
      const v16bf kb0 = frag_b(Ks + (t * 16) * 64,      64, lane);
      const v16bf kb1 = frag_b(Ks + (t * 16) * 64 + 32, 64, lane);
      sz = wmma_bf16(qa0, kb0, sz);
      sz = wmma_bf16(qa1, kb1, sz);
      sacc[t] = sz;
    }

    // online softmax: reduce across the 16 lanes holding each row
#pragma unroll
    for (int r8 = 0; r8 < 8; ++r8) {
      float v = fmaxf(fmaxf(sacc[0][r8], sacc[1][r8]),
                      fmaxf(sacc[2][r8], sacc[3][r8]));
#pragma unroll
      for (int off = 8; off > 0; off >>= 1) v = fmaxf(v, __shfl_xor(v, off, 32));
      const float mr    = fmaxf(mrun[r8], v);
      const float alpha = __expf(mrun[r8] - mr);
      mrun[r8] = mr;
      float ps = 0.f;
#pragma unroll
      for (int t = 0; t < 4; ++t) {
        const float pv = __expf(sacc[t][r8] - mr);
        ps += pv;
        Ps[(qr + r8 + hl * 8) * 64 + t * 16 + nl] = f2bf(pv);
      }
#pragma unroll
      for (int off = 8; off > 0; off >>= 1) ps += __shfl_xor(ps, off, 32);
      lrun[r8] = lrun[r8] * alpha + ps;
#pragma unroll
      for (int t = 0; t < 4; ++t) oacc[t][r8] *= alpha;
    }

    // wave-private Ps: wait for our own ds stores before the A-layout reads
    asm volatile("s_wait_dscnt 0" ::: "memory");

    // O += P @ V : A = P rows, B = V columns (Vt rows, contiguous in k)
    const v16bf pa0 = frag_a(Ps + qr * 64,      64, lane);
    const v16bf pa1 = frag_a(Ps + qr * 64 + 32, 64, lane);
#pragma unroll
    for (int t = 0; t < 4; ++t) {
      const v16bf vb0 = frag_b(Vt + (t * 16) * 64,      64, lane);
      const v16bf vb1 = frag_b(Vt + (t * 16) * 64 + 32, 64, lane);
      oacc[t] = wmma_bf16(pa0, vb0, oacc[t]);
      oacc[t] = wmma_bf16(pa1, vb1, oacc[t]);
    }
    __syncthreads();
  }

  // normalize and write bf16 into [B,S,D] (interleaved heads) for the Wo GEMM
#pragma unroll
  for (int r8 = 0; r8 < 8; ++r8) {
    const float inv = 1.0f / lrun[r8];
    const int row   = s0 + qr + r8 + hl * 8;
    const size_t base = ((size_t)b * 2048 + row) * 1024 + h * 64;
#pragma unroll
    for (int t = 0; t < 4; ++t) O[base + t * 16 + nl] = f2bf(oacc[t][r8] * inv);
  }
}

// ---------------------------------------------------------------------------
// Host launcher. Workspace (80 MB):
//   [0,32MB)   fp32 scratch for projection outputs (reused 3x), then bf16 attn out
//   [32,48MB)  Q head-major bf16   [48,64MB) K head-major   [64,80MB) V head-major
// ---------------------------------------------------------------------------
extern "C" void kernel_launch(void* const* d_in, const int* in_sizes, int n_in,
                              void* d_out, int out_size, void* d_ws, size_t ws_size,
                              hipStream_t stream) {
  const float* q  = (const float*)d_in[0];
  const float* k  = (const float*)d_in[1];
  const float* v  = (const float*)d_in[2];
  const float* Wq = (const float*)d_in[3];
  const float* Wk = (const float*)d_in[4];
  const float* Wv = (const float*)d_in[5];
  const float* Wo = (const float*)d_in[6];
  const float* bo = (const float*)d_in[7];
  const float* qg = (const float*)d_in[8];
  const float* qb = (const float*)d_in[9];
  const float* kg = (const float*)d_in[10];
  const float* kb = (const float*)d_in[11];

  const int M = 8192, N = 1024, K = 1024;
  char* ws = (char*)d_ws;
  float*          raw = (float*)ws;
  unsigned short* qh  = (unsigned short*)(ws + ((size_t)32 << 20));
  unsigned short* kh  = (unsigned short*)(ws + ((size_t)48 << 20));
  unsigned short* vh  = (unsigned short*)(ws + ((size_t)64 << 20));
  unsigned short* ah  = (unsigned short*)ws;  // attn out (bf16), reuses scratch

  const dim3 gG(N / 128, M / 128);   // 8 x 64 blocks, 256 threads
  const dim3 gA(32, 16, 4);          // (S/64, H, B), 128 threads

  // Q/K/V projections (+ LN for Q,K; scale 1/sqrt(1024) folded into Q)
  gemm_ln_proj<false, false><<<gG, 256, 0, stream>>>(q, Wq, nullptr, raw, M, N, K);
  ln_permute<true><<<M, 256, 0, stream>>>(raw, qg, qb, qh, 0.03125f);
  gemm_ln_proj<false, false><<<gG, 256, 0, stream>>>(k, Wk, nullptr, raw, M, N, K);
  ln_permute<true><<<M, 256, 0, stream>>>(raw, kg, kb, kh, 1.0f);
  gemm_ln_proj<false, false><<<gG, 256, 0, stream>>>(v, Wv, nullptr, raw, M, N, K);
  ln_permute<false><<<M, 256, 0, stream>>>(raw, nullptr, nullptr, vh, 1.0f);

  // fused attention (K staged by TDM)
  flash_attn<<<gA, 128, 0, stream>>>(qh, kh, vh, ah);

  // output projection + bias
  gemm_ln_proj<true, true><<<gG, 256, 0, stream>>>(ah, Wo, bo, (float*)d_out, M, N, K);
}
